// ConformerMultiHeadedSelfAttentionModule_23338852286565
// MI455X (gfx1250) — compile-verified, matrix operand order checked by reading, not run on hardware
//
#include <hip/hip_runtime.h>
#include <hip/hip_bf16.h>

typedef __attribute__((ext_vector_type(16))) __bf16 v16bf;
typedef __attribute__((ext_vector_type(8)))  float  v8f;

#define T_DIM 2048
#define B_DIM 4
#define C_DIM 512
#define H_DIM 8
#define DH    64
#define NROW  (T_DIM * B_DIM)
#define CLIPD 16
#define NREL  33

// ---------------------------------------------------------------------------
// CDNA5 helpers
// ---------------------------------------------------------------------------
static __device__ __forceinline__ v8f wmma_bf16(v16bf a, v16bf b, v8f c) {
  return __builtin_amdgcn_wmma_f32_16x16x32_bf16(
      false, a, false, b, (short)0, c, false, false);
}

// async global -> LDS copy of 16 bytes (tracked by ASYNCcnt)
static __device__ __forceinline__ void async_copy16(unsigned lds_off, const void* gaddr) {
  asm volatile("global_load_async_to_lds_b128 %0, %1, off"
               :: "v"(lds_off), "v"((unsigned long long)gaddr) : "memory");
}
static __device__ __forceinline__ void wait_async0() {
  asm volatile("s_wait_asynccnt 0x0" ::: "memory");
}
static __device__ __forceinline__ void wait_async2() {
  asm volatile("s_wait_asynccnt 0x2" ::: "memory");
}
static __device__ __forceinline__ void wait_async4() {
  asm volatile("s_wait_asynccnt 0x4" ::: "memory");
}

// Stage a 64(R) x 32(C) bf16 tile into LDS: 256 x 16B chunks, 128 thr x 2.
static __device__ __forceinline__ void stage_64x32(__bf16* dst, const __bf16* src,
                                                   int src_stride, int tid) {
  int c0 = tid, c1 = tid + 128;
  async_copy16((unsigned)(uintptr_t)(dst + (c0 >> 2) * 32 + (c0 & 3) * 8),
               src + (size_t)(c0 >> 2) * src_stride + (c0 & 3) * 8);
  async_copy16((unsigned)(uintptr_t)(dst + (c1 >> 2) * 32 + (c1 & 3) * 8),
               src + (size_t)(c1 >> 2) * src_stride + (c1 & 3) * 8);
}
// Stage a 32(R) x 64(C) bf16 tile into LDS.
static __device__ __forceinline__ void stage_32x64(__bf16* dst, const __bf16* src,
                                                   int src_stride, int tid) {
  int c0 = tid, c1 = tid + 128;
  async_copy16((unsigned)(uintptr_t)(dst + (c0 >> 3) * 64 + (c0 & 7) * 8),
               src + (size_t)(c0 >> 3) * src_stride + (c0 & 7) * 8);
  async_copy16((unsigned)(uintptr_t)(dst + (c1 >> 3) * 64 + (c1 & 7) * 8),
               src + (size_t)(c1 >> 3) * src_stride + (c1 & 7) * 8);
}

// Load a 16x32 bf16 fragment (A layout; B identical with row = n for an
// N x K row-major matrix).  ISA 7.12.2 K pattern -> two b128 loads per lane.
static __device__ __forceinline__ v16bf load_frag(const __bf16* tile, int stride, int lane) {
  int r = lane & 15;
  const __bf16* row = tile + (size_t)r * stride;
  int base = (lane & 16) ? 8 : 0;
  v16bf f;
#pragma unroll
  for (int p = 0; p < 8; ++p) {
    int k = (p < 4) ? (base + 2 * p) : (base + 16 + 2 * (p - 4));
    f[2 * p]     = row[k];
    f[2 * p + 1] = row[k + 1];
  }
  return f;
}

// ---------------------------------------------------------------------------
// Kernel 1: f32 -> bf16 conversion
// ---------------------------------------------------------------------------
__global__ void k_f32_to_bf16(const float* __restrict__ src, __bf16* __restrict__ dst, int n) {
  int i = blockIdx.x * blockDim.x + threadIdx.x;
  if (i < n) dst[i] = (__bf16)src[i];
}

// ---------------------------------------------------------------------------
// Kernel 2: LayerNorm (one 128-thread block per (t,b) row)
// ---------------------------------------------------------------------------
__global__ void k_layernorm(const float* __restrict__ x, const float* __restrict__ g,
                            const float* __restrict__ bta, __bf16* __restrict__ xn) {
  int row = blockIdx.x;
  int wave = threadIdx.x >> 5, lane = threadIdx.x & 31;
  const float4* xr = (const float4*)(x + (size_t)row * C_DIM);
  float4 v = xr[threadIdx.x];
  float s = v.x + v.y + v.z + v.w;
#pragma unroll
  for (int d = 1; d < 32; d <<= 1) s += __shfl_xor(s, d, 32);
  __shared__ float red[4], red2[4];
  if (lane == 0) red[wave] = s;
  __syncthreads();
  float mean = (red[0] + red[1] + red[2] + red[3]) * (1.0f / C_DIM);
  float ax = v.x - mean, ay = v.y - mean, az = v.z - mean, aw = v.w - mean;
  float sv = ax * ax + ay * ay + az * az + aw * aw;
#pragma unroll
  for (int d = 1; d < 32; d <<= 1) sv += __shfl_xor(sv, d, 32);
  if (lane == 0) red2[wave] = sv;
  __syncthreads();
  float var = (red2[0] + red2[1] + red2[2] + red2[3]) * (1.0f / C_DIM);
  float inv = rsqrtf(var + 1e-5f);
  int c0 = threadIdx.x * 4;
  __bf16* o = xn + (size_t)row * C_DIM + c0;
  o[0] = (__bf16)(ax * inv * g[c0 + 0] + bta[c0 + 0]);
  o[1] = (__bf16)(ay * inv * g[c0 + 1] + bta[c0 + 1]);
  o[2] = (__bf16)(az * inv * g[c0 + 2] + bta[c0 + 2]);
  o[3] = (__bf16)(aw * inv * g[c0 + 3] + bta[c0 + 3]);
}

// ---------------------------------------------------------------------------
// Kernel 3: QKV GEMM.  block = 128 thr = 4 waves; block tile 128(M) x 64(N);
// wave tile 32 x 64 (2 A frags x 4 B frags = 8 WMMA / k-step).  B tile staged
// once per block via async-LDS, double buffered.
// ---------------------------------------------------------------------------
__global__ void k_gemm_qkv(const __bf16* __restrict__ xn, const __bf16* __restrict__ w,
                           const float* __restrict__ bias,
                           __bf16* __restrict__ qo, __bf16* __restrict__ ko,
                           __bf16* __restrict__ vo) {
  __shared__ __bf16 bsh[2][64 * 32];
  int tid = threadIdx.x;
  int wave = tid >> 5, lane = tid & 31;
  int mblk = blockIdx.x * 128 + wave * 32;
  int nblk = blockIdx.y * 64;

  stage_64x32(&bsh[0][0], w + (size_t)nblk * C_DIM, C_DIM, tid);
  v8f acc[8] = {{}, {}, {}, {}, {}, {}, {}, {}};
  for (int ks = 0; ks < 16; ++ks) {
    int k0 = ks * 32;
    int cur = ks & 1;
    if (ks + 1 < 16) {
      stage_64x32(&bsh[cur ^ 1][0], w + (size_t)nblk * C_DIM + k0 + 32, C_DIM, tid);
      wait_async2();
    } else {
      wait_async0();
    }
    __syncthreads();
    v16bf a0 = load_frag(xn + (size_t)mblk * C_DIM + k0, C_DIM, lane);
    v16bf a1 = load_frag(xn + (size_t)(mblk + 16) * C_DIM + k0, C_DIM, lane);
    const __bf16* bb = &bsh[cur][0];
    v16bf b0 = load_frag(bb + 0 * 16 * 32, 32, lane);
    v16bf b1 = load_frag(bb + 1 * 16 * 32, 32, lane);
    v16bf b2 = load_frag(bb + 2 * 16 * 32, 32, lane);
    v16bf b3 = load_frag(bb + 3 * 16 * 32, 32, lane);
    acc[0] = wmma_bf16(a0, b0, acc[0]);
    acc[1] = wmma_bf16(a0, b1, acc[1]);
    acc[2] = wmma_bf16(a0, b2, acc[2]);
    acc[3] = wmma_bf16(a0, b3, acc[3]);
    acc[4] = wmma_bf16(a1, b0, acc[4]);
    acc[5] = wmma_bf16(a1, b1, acc[5]);
    acc[6] = wmma_bf16(a1, b2, acc[6]);
    acc[7] = wmma_bf16(a1, b3, acc[7]);
    __syncthreads();
  }

  int nn = lane & 15, mhi = (lane & 16) ? 8 : 0;
#pragma unroll
  for (int sub = 0; sub < 2; ++sub) {
#pragma unroll
    for (int nt = 0; nt < 4; ++nt) {
      int n = nblk + nt * 16 + nn;
      int which = n >> 9;
      int c = n & 511;
      int hh = c >> 6, dd = c & 63;
      float bs = bias[n];
#pragma unroll
      for (int j = 0; j < 8; ++j) {
        int m = mblk + sub * 16 + j + mhi;
        int t = m >> 2, bbk = m & 3;
        int bh = bbk * H_DIM + hh;
        float val = acc[sub * 4 + nt][j] + bs;
        if (which == 0)
          qo[((size_t)bh * T_DIM + t) * DH + dd] = (__bf16)(val * 0.125f);
        else if (which == 1)
          ko[((size_t)bh * T_DIM + t) * DH + dd] = (__bf16)val;
        else
          vo[((size_t)(bh * DH + dd)) * T_DIM + t] = (__bf16)val;   // transposed
      }
    }
  }
}

// ---------------------------------------------------------------------------
// Kernel 4: attention; K/V tiles async-staged to LDS, flash-style softmax.
//   grid = (T/64, B*H); block = 128 thr = 4 waves; wave owns 16 q rows.
// ---------------------------------------------------------------------------
__global__ void k_attention(const __bf16* __restrict__ qp, const __bf16* __restrict__ kp,
                            const __bf16* __restrict__ vt, const float* __restrict__ rel,
                            const unsigned char* __restrict__ maskp,
                            __bf16* __restrict__ attn) {
  int tid = threadIdx.x;
  int wave = tid >> 5, lane = tid & 31;
  int bh = blockIdx.y;
  int bb = bh / H_DIM, hh = bh % H_DIM;
  int qbase = blockIdx.x * 64 + wave * 16;

  const __bf16* qhead = qp + (size_t)bh * T_DIM * DH;
  const __bf16* khead = kp + (size_t)bh * T_DIM * DH;
  const __bf16* vhead = vt + (size_t)bh * DH * T_DIM;
  const unsigned char* mrow = maskp + (size_t)bb * T_DIM;

  __shared__ float  qr[4][16][NREL + 3];
  __shared__ __bf16 pst[4][16][32];
  __shared__ __bf16 ktile[2][32 * 64];   // keys x dh
  __shared__ __bf16 vtile[2][64 * 32];   // dh x keys

  // qr[m][r] = q_row(m) . rel_emb(r)
  for (int i = lane; i < 16 * NREL; i += 32) {
    int m = i / NREL, r = i % NREL;
    const __bf16* qrow = qhead + (size_t)(qbase + m) * DH;
    const float* er = rel + r * DH;
    float sum = 0.f;
#pragma unroll
    for (int d = 0; d < DH; ++d) sum += (float)qrow[d] * er[d];
    qr[wave][m][r] = sum;
  }

  v16bf qa0 = load_frag(qhead + (size_t)qbase * DH, DH, lane);
  v16bf qa1 = load_frag(qhead + (size_t)qbase * DH + 32, DH, lane);

  stage_32x64(&ktile[0][0], khead, DH, tid);
  stage_64x32(&vtile[0][0], vhead, T_DIM, tid);

  float rowm[8], rowl[8];
#pragma unroll
  for (int j = 0; j < 8; ++j) { rowm[j] = -3.0e38f; rowl[j] = 0.f; }
  v8f o0 = {}, o1 = {}, o2 = {}, o3 = {};

  int nn = lane & 15, mhi = (lane & 16) ? 8 : 0;

  for (int kb = 0; kb < T_DIM; kb += 32) {
    int cur = (kb >> 5) & 1;
    if (kb + 32 < T_DIM) {
      stage_32x64(&ktile[cur ^ 1][0], khead + (size_t)(kb + 32) * DH, DH, tid);
      stage_64x32(&vtile[cur ^ 1][0], vhead + kb + 32, T_DIM, tid);
      wait_async4();
    } else {
      wait_async0();
    }
    __syncthreads();                       // K/V tiles (and qr on iter 0) ready

    const __bf16* kt = &ktile[cur][0];
    const __bf16* vtb = &vtile[cur][0];

    // scores S(16x32) = Q @ K_tile^T
    v16bf kf0 = load_frag(kt, 64, lane);
    v16bf kf1 = load_frag(kt + 32, 64, lane);
    v16bf kf2 = load_frag(kt + 16 * 64, 64, lane);
    v16bf kf3 = load_frag(kt + 16 * 64 + 32, 64, lane);
    v8f s0 = {}, s1 = {};
    s0 = wmma_bf16(qa0, kf0, s0);
    s0 = wmma_bf16(qa1, kf1, s0);
    s1 = wmma_bf16(qa0, kf2, s1);
    s1 = wmma_bf16(qa1, kf3, s1);

    // relative-position bias + padding mask (C-fragment layout)
    int key0 = kb + nn, key1 = kb + 16 + nn;
    float msk0 = mrow[key0] ? -1e30f : 0.f;
    float msk1 = mrow[key1] ? -1e30f : 0.f;
#pragma unroll
    for (int j = 0; j < 8; ++j) {
      int m = j + mhi;
      int qrow = qbase + m;
      int r0 = min(max(key0 - qrow, -CLIPD), CLIPD) + CLIPD;
      int r1 = min(max(key1 - qrow, -CLIPD), CLIPD) + CLIPD;
      s0[j] += qr[wave][m][r0] + msk0;
      s1[j] += qr[wave][m][r1] + msk1;
    }

    // online softmax; one row = 16 lanes of a half-wave
#pragma unroll
    for (int j = 0; j < 8; ++j) {
      float mx = fmaxf(s0[j], s1[j]);
#pragma unroll
      for (int d = 1; d < 16; d <<= 1) mx = fmaxf(mx, __shfl_xor(mx, d, 32));
      float nm = fmaxf(rowm[j], mx);
      float corr = __expf(rowm[j] - nm);
      rowm[j] = nm;
      float p0 = __expf(s0[j] - nm);
      float p1 = __expf(s1[j] - nm);
      float ps = p0 + p1;
#pragma unroll
      for (int d = 1; d < 16; d <<= 1) ps += __shfl_xor(ps, d, 32);
      rowl[j] = rowl[j] * corr + ps;
      o0[j] *= corr; o1[j] *= corr; o2[j] *= corr; o3[j] *= corr;
      int m = j + mhi;
      pst[wave][m][nn]      = (__bf16)p0;
      pst[wave][m][nn + 16] = (__bf16)p1;
    }
    asm volatile("s_wait_dscnt 0x0" ::: "memory");
    __syncthreads();                       // P tile staged

    // O(16x64) += P(16x32) @ V(32x64)
    v16bf pf = load_frag(&pst[wave][0][0], 32, lane);
    v16bf vf0 = load_frag(vtb + 0 * 16 * 32, 32, lane);
    v16bf vf1 = load_frag(vtb + 1 * 16 * 32, 32, lane);
    v16bf vf2 = load_frag(vtb + 2 * 16 * 32, 32, lane);
    v16bf vf3 = load_frag(vtb + 3 * 16 * 32, 32, lane);
    o0 = wmma_bf16(pf, vf0, o0);
    o1 = wmma_bf16(pf, vf1, o1);
    o2 = wmma_bf16(pf, vf2, o2);
    o3 = wmma_bf16(pf, vf3, o3);
    __syncthreads();                       // all reads done before buffer reuse
  }

#pragma unroll
  for (int j = 0; j < 8; ++j) {
    int t = qbase + j + mhi;
    float inv = 1.0f / rowl[j];
    size_t roff = ((size_t)t * B_DIM + bb) * C_DIM + hh * DH;
    attn[roff + 0 * 16 + nn] = (__bf16)(o0[j] * inv);
    attn[roff + 1 * 16 + nn] = (__bf16)(o1[j] * inv);
    attn[roff + 2 * 16 + nn] = (__bf16)(o2[j] * inv);
    attn[roff + 3 * 16 + nn] = (__bf16)(o3[j] * inv);
  }
}

// ---------------------------------------------------------------------------
// Kernel 5: output projection; same 128x64 block tile as kernel 3, fp32 out
// ---------------------------------------------------------------------------
__global__ void k_gemm_out(const __bf16* __restrict__ a, const __bf16* __restrict__ w,
                           const float* __restrict__ bias, float* __restrict__ out) {
  __shared__ __bf16 bsh[2][64 * 32];
  int tid = threadIdx.x;
  int wave = tid >> 5, lane = tid & 31;
  int mblk = blockIdx.x * 128 + wave * 32;
  int nblk = blockIdx.y * 64;

  stage_64x32(&bsh[0][0], w + (size_t)nblk * C_DIM, C_DIM, tid);
  v8f acc[8] = {{}, {}, {}, {}, {}, {}, {}, {}};
  for (int ks = 0; ks < 16; ++ks) {
    int k0 = ks * 32;
    int cur = ks & 1;
    if (ks + 1 < 16) {
      stage_64x32(&bsh[cur ^ 1][0], w + (size_t)nblk * C_DIM + k0 + 32, C_DIM, tid);
      wait_async2();
    } else {
      wait_async0();
    }
    __syncthreads();
    v16bf a0 = load_frag(a + (size_t)mblk * C_DIM + k0, C_DIM, lane);
    v16bf a1 = load_frag(a + (size_t)(mblk + 16) * C_DIM + k0, C_DIM, lane);
    const __bf16* bb = &bsh[cur][0];
    v16bf b0 = load_frag(bb + 0 * 16 * 32, 32, lane);
    v16bf b1 = load_frag(bb + 1 * 16 * 32, 32, lane);
    v16bf b2 = load_frag(bb + 2 * 16 * 32, 32, lane);
    v16bf b3 = load_frag(bb + 3 * 16 * 32, 32, lane);
    acc[0] = wmma_bf16(a0, b0, acc[0]);
    acc[1] = wmma_bf16(a0, b1, acc[1]);
    acc[2] = wmma_bf16(a0, b2, acc[2]);
    acc[3] = wmma_bf16(a0, b3, acc[3]);
    acc[4] = wmma_bf16(a1, b0, acc[4]);
    acc[5] = wmma_bf16(a1, b1, acc[5]);
    acc[6] = wmma_bf16(a1, b2, acc[6]);
    acc[7] = wmma_bf16(a1, b3, acc[7]);
    __syncthreads();
  }

  int nn = lane & 15, mhi = (lane & 16) ? 8 : 0;
#pragma unroll
  for (int sub = 0; sub < 2; ++sub) {
#pragma unroll
    for (int j = 0; j < 8; ++j) {
      size_t r = (size_t)(mblk + sub * 16 + j + mhi) * C_DIM + nblk;
#pragma unroll
      for (int nt = 0; nt < 4; ++nt)
        out[r + nt * 16 + nn] = acc[sub * 4 + nt][j] + bias[nblk + nt * 16 + nn];
    }
  }
}

// ---------------------------------------------------------------------------
// Launch
// ---------------------------------------------------------------------------
extern "C" void kernel_launch(void* const* d_in, const int* in_sizes, int n_in,
                              void* d_out, int out_size, void* d_ws, size_t ws_size,
                              hipStream_t stream) {
  const float* x            = (const float*)d_in[0];
  const unsigned char* mask = (const unsigned char*)d_in[1];
  const float* ln_g         = (const float*)d_in[2];
  const float* ln_b         = (const float*)d_in[3];
  const float* w_qkv        = (const float*)d_in[4];
  const float* b_qkv        = (const float*)d_in[5];
  const float* w_out        = (const float*)d_in[6];
  const float* b_out        = (const float*)d_in[7];
  const float* rel          = (const float*)d_in[8];
  (void)in_sizes; (void)n_in; (void)out_size; (void)ws_size;

  char* ws = (char*)d_ws;
  size_t off = 0;
  __bf16* xn      = (__bf16*)(ws + off); off += (size_t)NROW * C_DIM * 2;
  __bf16* wqkv_bf = (__bf16*)(ws + off); off += (size_t)3 * C_DIM * C_DIM * 2;
  __bf16* wout_bf = (__bf16*)(ws + off); off += (size_t)C_DIM * C_DIM * 2;
  __bf16* qbuf    = (__bf16*)(ws + off); off += (size_t)B_DIM * H_DIM * T_DIM * DH * 2;
  __bf16* kbuf    = (__bf16*)(ws + off); off += (size_t)B_DIM * H_DIM * T_DIM * DH * 2;
  __bf16* vtbuf   = (__bf16*)(ws + off); off += (size_t)B_DIM * H_DIM * T_DIM * DH * 2;
  __bf16* attnbuf = (__bf16*)(ws + off); off += (size_t)NROW * C_DIM * 2;

  int nqkv = 3 * C_DIM * C_DIM, nout = C_DIM * C_DIM;
  k_f32_to_bf16<<<(nqkv + 255) / 256, 256, 0, stream>>>(w_qkv, wqkv_bf, nqkv);
  k_f32_to_bf16<<<(nout + 255) / 256, 256, 0, stream>>>(w_out, wout_bf, nout);
  k_layernorm<<<NROW, 128, 0, stream>>>(x, ln_g, ln_b, xn);
  k_gemm_qkv<<<dim3(NROW / 128, (3 * C_DIM) / 64), 128, 0, stream>>>(
      xn, wqkv_bf, b_qkv, qbuf, kbuf, vtbuf);
  k_attention<<<dim3(T_DIM / 64, B_DIM * H_DIM), 128, 0, stream>>>(
      qbuf, kbuf, vtbuf, rel, mask, attnbuf);
  k_gemm_out<<<dim3(NROW / 128, C_DIM / 64), 128, 0, stream>>>(
      attnbuf, wout_bf, b_out, (float*)d_out);
}
